// GAE_88656714924181
// MI455X (gfx1250) — matrix operand order repaired and checked by reading, not compiled, not run on hardware
//
#include <hip/hip_runtime.h>
#include <hip/hip_bf16.h>

typedef __attribute__((ext_vector_type(16))) _Float16 v16h;
typedef __attribute__((ext_vector_type(8)))  float    v8f;

#define NN    4096
#define CC    5
#define H0E   272      // 256 + 16 (ones/degree column tile)
#define DIN   512
#define SIDE_ 1024
#define H1_   128
#define A2LD  768      // DIN + H0 = 512 + 256
#define KCH   256      // K-chunk staged in LDS for the spmm kernels

// ---------------- WMMA helpers (16x16x32 f16 -> f32) ----------------

__device__ __forceinline__ v8f wmma_f16(v16h a, v16h b, v8f c) {
  return __builtin_amdgcn_wmma_f32_16x16x32_f16(false, a, false, b, (short)0, c, false, false);
}

// 16 contiguous halves -> v16h (16B-aligned)
__device__ __forceinline__ v16h ld16h(const _Float16* p) {
  union { uint4 u[2]; v16h h; } x;
  x.u[0] = *(const uint4*)p;
  x.u[1] = *(const uint4*)(p + 8);
  return x.h;
}
// p[0..7] and p[16..23] -> v16h (A-fragment pattern from f16 row-major)
__device__ __forceinline__ v16h ld8h_x2(const _Float16* p) {
  union { uint4 u[2]; v16h h; } x;
  x.u[0] = *(const uint4*)p;
  x.u[1] = *(const uint4*)(p + 16);
  return x.h;
}

// A fragment (16x32 f16) from f16 row-major A[lda], rows m0.., K at k0
__device__ __forceinline__ v16h a_frag_h(const _Float16* A, int lda, int m0, int k0, int lane) {
  int m  = m0 + (lane & 15);
  int kb = k0 + ((lane >> 4) << 4);     // lanes>=16: +16
  return ld16h(A + (size_t)m * lda + kb);
}
// A fragment from f32 row-major source (convert)
__device__ __forceinline__ v16h a_frag_f32row(const float* Arow, int k0, int lane) {
  int kb = k0 + ((lane >> 4) << 4);
  const float* p = Arow + kb;
  v16h r;
#pragma unroll
  for (int e = 0; e < 8; ++e) r[e] = (_Float16)p[e];
#pragma unroll
  for (int e = 0; e < 8; ++e) r[8 + e] = (_Float16)p[8 + e];
  return r;
}
// B fragment (32x16 f16) from f16 "B^T" row-major Bt[n][k], row stride ldk
__device__ __forceinline__ v16h b_frag_h(const _Float16* Bt, int ldk, int n0, int k0, int lane) {
  int n = n0 + (lane & 15);
  int k = k0 + ((lane >> 4) << 4);      // lanes>=16: +16
  return ld16h(Bt + (size_t)n * ldk + k);
}
// B fragment from f32 "B^T" row-major W[n][k] (convert)
__device__ __forceinline__ v16h b_frag_f32(const float* Bt, int ldk, int n0, int k0, int lane) {
  int n = n0 + (lane & 15);
  int k = k0 + ((lane >> 4) << 4);
  const float* p = Bt + (size_t)n * ldk + k;
  v16h r;
#pragma unroll
  for (int e = 0; e < 16; ++e) r[e] = (_Float16)p[e];
  return r;
}
// B fragment from f32 row-major P[k][n] (transposed-on-the-fly, tiny matrix)
__device__ __forceinline__ v16h b_frag_f32T(const float* P, int ld, int n0, int k0, int lane) {
  int n  = n0 + (lane & 15);
  int ks = k0 + ((lane >> 4) << 4);
  v16h r;
#pragma unroll
  for (int e = 0; e < 16; ++e) r[e] = (_Float16)P[(size_t)(ks + e) * ld + n];
  return r;
}
// build 0/1 f16 A fragment from 16 packed code bytes, channel c+1
__device__ __forceinline__ v16h code_frag(unsigned long long w0, unsigned long long w1, unsigned cval) {
  v16h r;
#pragma unroll
  for (int e = 0; e < 8; ++e) {
    unsigned b = (unsigned)(w0 >> (8 * e)) & 0xFFu;
    r[e] = (b == cval) ? (_Float16)1.0f : (_Float16)0.0f;
  }
#pragma unroll
  for (int e = 0; e < 8; ++e) {
    unsigned b = (unsigned)(w1 >> (8 * e)) & 0xFFu;
    r[8 + e] = (b == cval) ? (_Float16)1.0f : (_Float16)0.0f;
  }
  return r;
}

// ---------------- Tensor Data Mover staging (gfx1250) ----------------
#if __has_builtin(__builtin_amdgcn_tensor_load_to_lds) && __has_builtin(__builtin_amdgcn_s_wait_tensorcnt)
#define GAE_TDM 1
typedef unsigned int v4u __attribute__((ext_vector_type(4)));
typedef int          v4i __attribute__((ext_vector_type(4)));
typedef int          v8i __attribute__((ext_vector_type(8)));

// 2D tile load: tile_h rows x tile_w elements (2 bytes each), row stride row_stride elems.
// Descriptor packing per CDNA5 ISA 8.3/8.4 (D# group0/group1).
__device__ __forceinline__ void tdm_load_2d(unsigned lds_off, const void* gaddr,
                                            unsigned tile_w, unsigned tile_h,
                                            unsigned tensor_w, unsigned row_stride) {
  unsigned long long ga = (unsigned long long)(uintptr_t)gaddr;
  v4u g0;
  g0[0] = 1u;                                               // count=1, user descriptor
  g0[1] = lds_off;                                          // lds_addr (bytes)
  g0[2] = (unsigned)ga;                                     // global_addr[31:0]
  g0[3] = (unsigned)((ga >> 32) & 0x01FFFFFFu) | (2u << 30);// global_addr[56:32] | type=2
  v8i g1;
  g1[0] = (int)(1u << 16);                                  // data_size=1 (2 bytes)
  g1[1] = (int)((tensor_w & 0xFFFFu) << 16);                // tensor_dim0[15:0]
  g1[2] = (int)((tensor_w >> 16) & 0xFFFFu);                // tensor_dim0[31:16] | td1[15:0]=0
  g1[3] = (int)(1u | (tile_w << 16));                       // td1[31:16]=1 (dim1=65536) | tile_dim0
  g1[4] = (int)(tile_h & 0xFFFFu);                          // tile_dim1 | tile_dim2=0
  g1[5] = (int)row_stride;                                  // tensor_dim0_stride[31:0]
  g1[6] = 0;                                                // tds0[47:32] | tds1[15:0]
  g1[7] = 0;                                                // tds1[47:16]
  v4i z = {0, 0, 0, 0};
#if defined(__clang_major__) && (__clang_major__ >= 23)
  v8i z8 = {0, 0, 0, 0, 0, 0, 0, 0};
  __builtin_amdgcn_tensor_load_to_lds(g0, g1, z, z, z8, 0);
#else
  __builtin_amdgcn_tensor_load_to_lds(g0, g1, z, z, 0);
#endif
}
#else
#define GAE_TDM 0
#endif

// ---------------- Stage 0: cumsum-over-c + gather + transpose -> xT[c][272][4096] f16 ----------------
__global__ void gae_prep_x(const float* __restrict__ w, const int* __restrict__ idx,
                           _Float16* __restrict__ xT) {
  __shared__ _Float16 lds[256 * 32];
  int t   = threadIdx.x;        // 256
  int i0  = blockIdx.x * 32;
  int il  = t >> 3;             // 0..31  local row
  int seg = t & 7;              // 0..7   h segment of 32
  int row = idx[i0 + il];
  const float* wr = w + (size_t)row * 256 + seg * 32;
  float acc[32];
#pragma unroll
  for (int k = 0; k < 32; ++k) acc[k] = 0.f;

  for (int c = 0; c < CC; ++c) {
    const float* p = wr + (size_t)c * NN * 256;
#pragma unroll
    for (int k = 0; k < 32; ++k) acc[k] += p[k];
#pragma unroll
    for (int k = 0; k < 32; ++k) lds[(seg * 32 + k) * 32 + il] = (_Float16)acc[k];
    __syncthreads();
    int ilw = t & 31, hb = t >> 5;   // write h-major, lanes along i
    _Float16* dst = xT + (size_t)c * H0E * NN + i0 + ilw;
    for (int h = hb; h < 256; h += 8) dst[(size_t)h * NN] = lds[h * 32 + ilw];
    __syncthreads();
  }
  if (t < 32) {                      // ones column at h=256, zeros at 257..271
    for (int c = 0; c < CC; ++c) {
      _Float16* dst = xT + (size_t)c * H0E * NN + (size_t)256 * NN + i0 + t;
      dst[0] = (_Float16)1.0f;
      for (int h = 1; h < 16; ++h) dst[(size_t)h * NN] = (_Float16)0.0f;
    }
  }
}

// ---------------- Stage 1: compress gathered adjacency to u8 code[i][j] ----------------
__global__ void gae_compress(const float* __restrict__ adj, const int* __restrict__ u,
                             const int* __restrict__ v, unsigned char* __restrict__ code) {
  int tid = blockIdx.x * blockDim.x + threadIdx.x;   // 4096*1024 threads, 4 j each
  int i  = tid >> 10;
  int j4 = (tid & 1023) << 2;
  int ur = u[i];
  const float* base = adj + (size_t)ur * NN;
  unsigned char c0 = 0, c1 = 0, c2 = 0, c3 = 0;
  int vj0 = v[j4], vj1 = v[j4 + 1], vj2 = v[j4 + 2], vj3 = v[j4 + 3];
#pragma unroll
  for (int c = 0; c < CC; ++c) {
    const float* p = base + (size_t)c * NN * NN;
    if (p[vj0] != 0.f) c0 = (unsigned char)(c + 1);
    if (p[vj1] != 0.f) c1 = (unsigned char)(c + 1);
    if (p[vj2] != 0.f) c2 = (unsigned char)(c + 1);
    if (p[vj3] != 0.f) c3 = (unsigned char)(c + 1);
  }
  uchar4 o; o.x = c0; o.y = c1; o.z = c2; o.w = c3;
  *(uchar4*)(code + (size_t)i * NN + j4) = o;
}

// ---------------- Stage 2: Y[row][h] = sum_c sum_k (code==c+1) * Xt[c][h][k] ----------------
// Block = 8 waves sharing one h-tile; B panels staged in LDS (TDM when available).
// KONE=1: row-major code (u_next); KONE=0: k-strided (v_next, implicit transpose).
template <bool KONE>
__global__ __launch_bounds__(256) void gae_spmm(const unsigned char* __restrict__ code,
                                                long rowStride, long kStride,
                                                const _Float16* __restrict__ Xt,
                                                float* __restrict__ Y) {
  __shared__ _Float16 panel[CC][16][KCH];   // 40KB
  int tid  = threadIdx.x;
  int lane = tid & 31;
  int wave = tid >> 5;          // 0..7
  int h0 = blockIdx.x * 16;     // 0..271
  int r0 = blockIdx.y * 128 + wave * 16;
  int m  = r0 + (lane & 15);
  int kboff = (lane >> 4) << 4;
  const unsigned char* base = code + (size_t)m * rowStride;
  int nloc = lane & 15;
  int kloc = (lane >> 4) << 4;
  v8f acc = {};

  for (int kc = 0; kc < NN; kc += KCH) {
#if GAE_TDM
    if (wave == 0) {
      if (lane == 0) {
#pragma unroll
        for (int c = 0; c < CC; ++c)
          tdm_load_2d((unsigned)(uintptr_t)&panel[c][0][0],
                      Xt + (size_t)c * H0E * NN + (size_t)h0 * NN + kc,
                      KCH, 16, NN, NN);
      }
      __builtin_amdgcn_s_wait_tensorcnt(0);
    }
    __syncthreads();
#else
    for (int idx = tid; idx < CC * 16 * (KCH / 8); idx += 256) {
      int col8 = idx % (KCH / 8);
      int rc   = idx / (KCH / 8);
      int row  = rc & 15, c = rc >> 4;
      uint4 d = *(const uint4*)(Xt + (size_t)c * H0E * NN + (size_t)(h0 + row) * NN + kc + col8 * 8);
      *(uint4*)&panel[c][row][col8 * 8] = d;
    }
    __syncthreads();
#endif
    for (int k0 = 0; k0 < KCH; k0 += 32) {
      int kb = kc + k0 + kboff;
      unsigned long long w0, w1;
      if (KONE) {
        const unsigned long long* p = (const unsigned long long*)(base + kb);
        w0 = p[0];
        w1 = p[1];
        __builtin_prefetch(base + kb + 256, 0, 1);
      } else {
        w0 = 0; w1 = 0;
#pragma unroll
        for (int e = 0; e < 8; ++e)
          w0 |= (unsigned long long)base[(size_t)(kb + e) * kStride] << (8 * e);
#pragma unroll
        for (int e = 0; e < 8; ++e)
          w1 |= (unsigned long long)base[(size_t)(kb + 8 + e) * kStride] << (8 * e);
      }
#pragma unroll
      for (int c = 0; c < CC; ++c) {
        v16h a = code_frag(w0, w1, (unsigned)(c + 1));
        v16h b = ld16h(&panel[c][nloc][k0 + kloc]);   // ds_load_b128 x2
        acc = wmma_f16(a, b, acc);
      }
    }
    __syncthreads();
  }
  int n = lane & 15, mo = (lane >> 4) * 8;
#pragma unroll
  for (int r = 0; r < 8; ++r) Y[(size_t)(r0 + r + mo) * H0E + h0 + n] = acc[r];
}

// ---------------- Stage 3: feature MLP1: relu(feats[idx] @ W^T + b) -> f16 into A2[:,256:768] ----------------
__global__ void gae_feat_gemm(const float* __restrict__ feats, const int* __restrict__ idx,
                              const float* __restrict__ W, const float* __restrict__ bias,
                              _Float16* __restrict__ A2) {
  int lane = threadIdx.x;
  int n0 = blockIdx.x * 16;     // 0..511
  int i0 = blockIdx.y * 16;
  int row = idx[i0 + (lane & 15)];
  const float* Arow = feats + (size_t)row * SIDE_;
  v8f acc = {};
  for (int k0 = 0; k0 < SIDE_; k0 += 32) {
    v16h a = a_frag_f32row(Arow, k0, lane);
    v16h b = b_frag_f32(W, SIDE_, n0, k0, lane);
    acc = wmma_f16(a, b, acc);
  }
  int n = lane & 15, mo = (lane >> 4) * 8;
  float bn = bias[n0 + n];
#pragma unroll
  for (int r = 0; r < 8; ++r) {
    float x = fmaxf(acc[r] + bn, 0.f);
    A2[(size_t)(i0 + r + mo) * A2LD + 256 + n0 + n] = (_Float16)x;
  }
}

// ---------------- Stage 4: z scaling: relu(Y * 1/deg) -> f16 into A2[:,0:256] ----------------
__global__ void gae_scale_z(const float* __restrict__ Yraw, const float* __restrict__ degsrc,
                            _Float16* __restrict__ A2) {
  int tid = blockIdx.x * blockDim.x + threadIdx.x;   // 4096*64
  int i = tid >> 6;
  int h4 = (tid & 63) << 2;
  float d = fabsf(degsrc[(size_t)i * H0E + 256]);
  float s = (d != 0.f) ? 1.f / d : 0.f;              // 1/0 = inf -> 0
  const float* y = Yraw + (size_t)i * H0E + h4;
  _Float16* o = A2 + (size_t)i * A2LD + h4;
#pragma unroll
  for (int q = 0; q < 4; ++q) o[q] = (_Float16)fmaxf(y[q] * s, 0.f);
}

// ---------------- Stage 5: h = relu(A2 @ W2^T) -> f16 [4096][128] ----------------
__global__ void gae_h_gemm(const _Float16* __restrict__ A2, const float* __restrict__ W2,
                           _Float16* __restrict__ Hout) {
  int lane = threadIdx.x;
  int n0 = blockIdx.x * 16;     // 0..127
  int i0 = blockIdx.y * 16;
  v8f acc = {};
  for (int k0 = 0; k0 < A2LD; k0 += 32) {
    v16h a = a_frag_h(A2, A2LD, i0, k0, lane);
    v16h b = b_frag_f32(W2, A2LD, n0, k0, lane);
    acc = wmma_f16(a, b, acc);
  }
  int n = lane & 15, mo = (lane >> 4) * 8;
#pragma unroll
  for (int r = 0; r < 8; ++r)
    Hout[(size_t)(i0 + r + mo) * H1_ + n0 + n] = (_Float16)fmaxf(acc[r], 0.f);
}

// ---------------- Stage 6: t[b] = u_h @ P[b] -> f16 [2][4096][128] ----------------
__global__ void gae_t_gemm(const _Float16* __restrict__ uh, const float* __restrict__ P,
                           _Float16* __restrict__ T) {
  int lane = threadIdx.x;
  int n0 = blockIdx.x * 16;
  int i0 = blockIdx.y * 16;
  int b  = blockIdx.z;
  const float* Pb = P + (size_t)b * H1_ * H1_;
  v8f acc = {};
  for (int k0 = 0; k0 < H1_; k0 += 32) {
    v16h a = a_frag_h(uh, H1_, i0, k0, lane);
    v16h bf = b_frag_f32T(Pb, H1_, n0, k0, lane);
    acc = wmma_f16(a, bf, acc);
  }
  int n = lane & 15, mo = (lane >> 4) * 8;
  _Float16* Tb = T + (size_t)b * NN * H1_;
#pragma unroll
  for (int r = 0; r < 8; ++r) Tb[(size_t)(i0 + r + mo) * H1_ + n0 + n] = (_Float16)acc[r];
}

// ---------------- Stage 7: fused bilinear + softmax + m_hat + loss/rmse reductions ----------------
__global__ void gae_final(const _Float16* __restrict__ T,    // [2][4096][128]
                          const _Float16* __restrict__ vh,   // [4096][128]
                          const unsigned char* __restrict__ code,
                          const float* __restrict__ acoef,   // [2][5]
                          float* __restrict__ mhat, float* __restrict__ accums) {
  int lane = threadIdx.x;
  int j0 = blockIdx.x * 16;
  int i0 = blockIdx.y * 16;
  v8f g0 = {}, g1 = {};
  for (int k0 = 0; k0 < H1_; k0 += 32) {
    v16h b  = b_frag_h(vh, H1_, j0, k0, lane);
    v16h a0 = a_frag_h(T, H1_, i0, k0, lane);
    v16h a1 = a_frag_h(T + (size_t)NN * H1_, H1_, i0, k0, lane);
    g0 = wmma_f16(a0, b, g0);
    g1 = wmma_f16(a1, b, g1);
  }
  float a0c[CC], a1c[CC];
#pragma unroll
  for (int c = 0; c < CC; ++c) { a0c[c] = acoef[c]; a1c[c] = acoef[CC + c]; }

  int n = lane & 15, mo = (lane >> 4) * 8;
  float lsum = 0.f, sesum = 0.f, cnt = 0.f;
#pragma unroll
  for (int r = 0; r < 8; ++r) {
    int i = i0 + r + mo, j = j0 + n;
    float x0 = g0[r], x1 = g1[r];
    float lg[CC], mx = -3.4e38f;
#pragma unroll
    for (int c = 0; c < CC; ++c) { lg[c] = a0c[c] * x0 + a1c[c] * x1; mx = fmaxf(mx, lg[c]); }
    float s = 0.f;
#pragma unroll
    for (int c = 0; c < CC; ++c) s += __expf(lg[c] - mx);
    float lse = mx + __logf(s);
    float inv = 1.f / s;
    float mh = 0.f;
#pragma unroll
    for (int c = 0; c < CC; ++c) mh += (float)(c + 1) * __expf(lg[c] - mx) * inv;
    mhat[(size_t)i * NN + j] = mh;
    int cb = code[(size_t)i * NN + j];
    if (cb) {
      lsum += (lse - lg[cb - 1]);           // -(m*logp)
      cnt  += 1.f;
      float d = mh - (float)cb;             // target = argmax+1 = cb
      sesum += d * d;
    }
  }
#pragma unroll
  for (int off = 16; off > 0; off >>= 1) {
    lsum  += __shfl_down(lsum, off, 32);
    sesum += __shfl_down(sesum, off, 32);
    cnt   += __shfl_down(cnt, off, 32);
  }
  if (lane == 0) {
    atomicAdd(&accums[0], lsum);
    atomicAdd(&accums[1], sesum);
    atomicAdd(&accums[2], cnt);
  }
}

__global__ void gae_zero_acc(float* accums) {
  if (threadIdx.x < 8) accums[threadIdx.x] = 0.f;
}

__global__ void gae_finalize(const float* __restrict__ accums, float* __restrict__ out_tail) {
  float n = fmaxf(accums[2], 1.f);
  out_tail[0] = accums[0] / n;
  out_tail[1] = sqrtf(accums[1] / n);
}

// ---------------- host launch ----------------

static inline size_t alup(size_t x) { return (x + 255) & ~(size_t)255; }

extern "C" void kernel_launch(void* const* d_in, const int* in_sizes, int n_in,
                              void* d_out, int out_size, void* d_ws, size_t ws_size,
                              hipStream_t stream) {
  const int*   u   = (const int*)d_in[0];
  const int*   v   = (const int*)d_in[1];
  const float* adj = (const float*)d_in[3];
  const float* uf  = (const float*)d_in[4];
  const float* vf  = (const float*)d_in[5];
  const float* uw  = (const float*)d_in[6];
  const float* vw  = (const float*)d_in[7];
  const float* Wu1 = (const float*)d_in[8];
  const float* bu1 = (const float*)d_in[9];
  const float* Wv1 = (const float*)d_in[10];
  const float* bv1 = (const float*)d_in[11];
  const float* Wu2 = (const float*)d_in[12];
  const float* Wv2 = (const float*)d_in[13];
  const float* P   = (const float*)d_in[14];
  const float* ac  = (const float*)d_in[15];
  float* out = (float*)d_out;

  char* ws = (char*)d_ws;
  size_t off = 0;
  _Float16* uxT  = (_Float16*)(ws + off); off = alup(off + (size_t)CC * H0E * NN * 2);
  _Float16* vxT  = (_Float16*)(ws + off); off = alup(off + (size_t)CC * H0E * NN * 2);
  unsigned char* code = (unsigned char*)(ws + off); off = alup(off + (size_t)NN * NN);
  float* unraw = (float*)(ws + off); off = alup(off + (size_t)NN * H0E * 4);
  float* vnraw = (float*)(ws + off); off = alup(off + (size_t)NN * H0E * 4);
  _Float16* A2u = (_Float16*)(ws + off); off = alup(off + (size_t)NN * A2LD * 2);
  _Float16* A2v = (_Float16*)(ws + off); off = alup(off + (size_t)NN * A2LD * 2);
  _Float16* uh  = (_Float16*)(ws + off); off = alup(off + (size_t)NN * H1_ * 2);
  _Float16* vh  = (_Float16*)(ws + off); off = alup(off + (size_t)NN * H1_ * 2);
  _Float16* T   = (_Float16*)(ws + off); off = alup(off + (size_t)2 * NN * H1_ * 2);
  float* accums = (float*)(ws + off); off = alup(off + 256);

  hipLaunchKernelGGL(gae_zero_acc, dim3(1), dim3(32), 0, stream, accums);

  // cumsum+gather+transpose
  hipLaunchKernelGGL(gae_prep_x, dim3(NN / 32), dim3(256), 0, stream, uw, u, uxT);
  hipLaunchKernelGGL(gae_prep_x, dim3(NN / 32), dim3(256), 0, stream, vw, v, vxT);

  // compress adjacency
  hipLaunchKernelGGL(gae_compress, dim3((NN * 1024) / 256), dim3(256), 0, stream, adj, u, v, code);

  // message passing GEMMs (+degree columns); 8 waves/block share LDS B panel
  hipLaunchKernelGGL((gae_spmm<true>),  dim3(H0E / 16, NN / 128), dim3(256), 0, stream,
                     code, (long)NN, (long)1, vxT, unraw);
  hipLaunchKernelGGL((gae_spmm<false>), dim3(H0E / 16, NN / 128), dim3(256), 0, stream,
                     code, (long)1, (long)NN, uxT, vnraw);

  // feature MLP1 -> A2[:,256:768]
  hipLaunchKernelGGL(gae_feat_gemm, dim3(DIN / 16, NN / 16), dim3(32), 0, stream, uf, u, Wu1, bu1, A2u);
  hipLaunchKernelGGL(gae_feat_gemm, dim3(DIN / 16, NN / 16), dim3(32), 0, stream, vf, v, Wv1, bv1, A2v);

  // degree scaling -> A2[:,0:256]   (u scaled by du from v-side GEMM, and vice versa)
  hipLaunchKernelGGL(gae_scale_z, dim3((NN * 64) / 256), dim3(256), 0, stream, unraw, vnraw, A2u);
  hipLaunchKernelGGL(gae_scale_z, dim3((NN * 64) / 256), dim3(256), 0, stream, vnraw, unraw, A2v);

  // MLP2 -> u_h, v_h (f16)
  hipLaunchKernelGGL(gae_h_gemm, dim3(H1_ / 16, NN / 16), dim3(32), 0, stream, A2u, Wu2, uh);
  hipLaunchKernelGGL(gae_h_gemm, dim3(H1_ / 16, NN / 16), dim3(32), 0, stream, A2v, Wv2, vh);

  // t[b] = u_h @ P[b]
  hipLaunchKernelGGL(gae_t_gemm, dim3(H1_ / 16, NN / 16, 2), dim3(32), 0, stream, uh, P, T);

  // fused bilinear + softmax + reductions
  hipLaunchKernelGGL(gae_final, dim3(NN / 16, NN / 16), dim3(32), 0, stream, T, vh, code, ac, out, accums);

  // scalars
  hipLaunchKernelGGL(gae_finalize, dim3(1), dim3(1), 0, stream, accums, out + (size_t)NN * NN);

  (void)in_sizes; (void)n_in; (void)out_size; (void)ws_size;
}
